// Attention_19413252178452
// MI455X (gfx1250) — matrix operand order repaired and checked by reading, not compiled, not run on hardware
//
#include <hip/hip_runtime.h>
#include <hip/hip_bf16.h>
#include <stdint.h>

// ---------------------------------------------------------------------------
// Problem constants (match reference)
// ---------------------------------------------------------------------------
#define B_   2
#define N_   2048
#define DIM_ 2048
#define H_   16
#define HD_  128
#define EPS_ 1e-5f

typedef __bf16 bf16_t;
typedef __attribute__((ext_vector_type(16))) __bf16 v16bf;
typedef __attribute__((ext_vector_type(4)))  __bf16 v4bf;
typedef __attribute__((ext_vector_type(8)))  float  v8f;

union Frag16 { v16bf v; uint4 q[2]; };   // 32B: one WMMA bf16 A/B operand
union F8     { v8f   v; float f[8];  };  // 32B: one WMMA f32  C/D operand

static __device__ __forceinline__ v8f wmma_bf16(v16bf a, v16bf b, v8f c) {
  // D = A(16x32) * B(32x16) + C(16x16), f32 accumulate
  return __builtin_amdgcn_wmma_f32_16x16x32_bf16(
      /*neg_a=*/false, a, /*neg_b=*/false, b,
      /*c_mod=*/(short)0, c, /*reuse_a=*/false, /*reuse_b=*/false);
}

// ---------------------------------------------------------------------------
// Tensor Data Mover support (gfx1250).
//  - clang-22 (ROCm 7.2):  5 args (g0, g1, g2, g3, cpol)
//  - clang-23 (amdgpu-toolchain): 6 args (g0, g1, g2, g3, g4, cpol)
// Guarded so absence of the builtin falls back to direct global loads.
// ---------------------------------------------------------------------------
#if defined(__has_builtin)
#if __has_builtin(__builtin_amdgcn_tensor_load_to_lds) && \
    __has_builtin(__builtin_amdgcn_s_wait_tensorcnt)
#define USE_TDM 1
#endif
#endif
#ifndef USE_TDM
#define USE_TDM 0
#endif

#if USE_TDM
typedef __attribute__((ext_vector_type(4))) unsigned int v4u;
typedef __attribute__((ext_vector_type(8))) int v8i;
typedef __attribute__((ext_vector_type(4))) int v4i;

// Issue a 2D TDM tile load (2-byte elements) with LDS row padding.
//  tensor_d0/stride0 : contiguous dimension length / row pitch (elements)
//  tile_d0 x tile_d1 : tile size (elements x rows)
//  pad codes per D# group1: interval dwords = 2<<code, amount dwords = code+1
static __device__ __forceinline__ void tdm_load_2d(
    const void* gptr, unsigned lds_byte_addr, unsigned tensor_d0,
    unsigned tensor_d1, unsigned tile_d0, unsigned tile_d1,
    unsigned long long stride0, unsigned pad_interval_code,
    unsigned pad_amount_code) {
  unsigned long long ga = (unsigned long long)(uintptr_t)gptr;
  v4u g0;
  g0.x = 1u;                                   // count=1, user descriptor
  g0.y = lds_byte_addr;                        // LDS destination (bytes)
  g0.z = (unsigned)(ga & 0xffffffffu);         // global_addr[31:0]
  g0.w = (unsigned)((ga >> 32) & 0x01ffffffu)  // global_addr[56:32]
         | (2u << 30);                         // type = 2 ("image")
  v8i g1;
  g1[0] = (int)((1u << 16)                     // data_size = 2 bytes
                | (1u << 20)                   // pad_enable
                | (pad_interval_code << 22) | (pad_amount_code << 25));
  g1[1] = (int)((tensor_d0 & 0xffffu) << 16);  // tensor_dim0[15:0]
  g1[2] = (int)((tensor_d0 >> 16) | ((tensor_d1 & 0xffffu) << 16));
  g1[3] = (int)((tensor_d1 >> 16) | (tile_d0 << 16));
  g1[4] = (int)tile_d1;                        // tile_dim2 = 0 (unused)
  g1[5] = (int)(stride0 & 0xffffffffull);      // tensor_dim0_stride[31:0]
  g1[6] = (int)((unsigned)(stride0 >> 32) & 0xffffu);  // stride0 hi, stride1=0
  g1[7] = 0;
  v4i gz4 = {0, 0, 0, 0};
#if __clang_major__ >= 23
  v8i gz8 = {0, 0, 0, 0, 0, 0, 0, 0};
  __builtin_amdgcn_tensor_load_to_lds(g0, g1, gz4, gz4, gz8, 0);
#else
  __builtin_amdgcn_tensor_load_to_lds(g0, g1, gz4, gz4, 0);
#endif
}
#endif

// ---------------------------------------------------------------------------
// Generic GEMM: C[M x NTOT] = A[M x 2048] * W[2048 x NTOT]
//  - A is f32 (converted to bf16 on LDS store) or already bf16 (ABF=true)
//  - W is f32, converted + transposed into LDS
//  - 256 threads = 8 waves; block tile 128x128, BK=32; wave tile 32x64
//  - LDS row stride padded to 40 bf16 -> conflict-free b128 fragment loads
// ---------------------------------------------------------------------------
template <int NTOT, bool ABF>
__global__ __launch_bounds__(256) void gemm_kernel(const void* __restrict__ Ain,
                                                   const float* __restrict__ W,
                                                   float* __restrict__ Cout) {
  constexpr int K = DIM_;
  constexpr int LD = 40;                       // padded row stride (bf16 elems)
  __shared__ bf16_t As[128 * LD];              // 128 rows  x 32 k
  __shared__ bf16_t Bs[128 * LD];              // 128 cols  x 32 k (transposed)

  const int t    = threadIdx.x;
  const int lane = t & 31;
  const int wid  = t >> 5;
  const int lo   = lane & 15;                  // row/col within 16
  const int hi   = lane >> 4;                  // half-wave select
  const int wm   = wid & 3;                    // 4 wave-rows  x 32
  const int wn   = wid >> 2;                   // 2 wave-cols  x 64
  const int rowBase = blockIdx.y * 128;
  const int colBase = blockIdx.x * 128;

  v8f zero = {};
  F8 acc[2][4];
#pragma unroll
  for (int fm = 0; fm < 2; ++fm)
#pragma unroll
    for (int fn = 0; fn < 4; ++fn) acc[fm][fn].v = zero;

  const float*  Af = (const float*)Ain;
  const bf16_t* Ab = (const bf16_t*)Ain;

  for (int k0 = 0; k0 < K; k0 += 32) {
    // ---- cooperative tile load: A (128x32) ----
    if (ABF) {
#pragma unroll
      for (int i = 0; i < 2; ++i) {            // 512 chunks of 8 bf16
        int idx = t + i * 256;
        int row = idx >> 2, c8 = idx & 3;
        *(uint4*)(As + row * LD + c8 * 8) =
            *(const uint4*)(Ab + (size_t)(rowBase + row) * K + k0 + c8 * 8);
      }
    } else {
#pragma unroll
      for (int i = 0; i < 4; ++i) {            // 1024 float4 chunks
        int idx = t + i * 256;
        int row = idx >> 3, c4 = idx & 7;
        float4 f = *(const float4*)(Af + (size_t)(rowBase + row) * K + k0 + c4 * 4);
        v4bf v = {(__bf16)f.x, (__bf16)f.y, (__bf16)f.z, (__bf16)f.w};
        *(v4bf*)(As + row * LD + c4 * 4) = v;
      }
    }
    // ---- cooperative tile load: W (32 x 128) -> transposed bf16 ----
#pragma unroll
    for (int i = 0; i < 4; ++i) {
      int idx = t + i * 256;
      int kr = idx >> 5, c4 = idx & 31;
      int col = c4 * 4;
      float4 f = *(const float4*)(W + (size_t)(k0 + kr) * NTOT + colBase + col);
      Bs[(col + 0) * LD + kr] = (__bf16)f.x;
      Bs[(col + 1) * LD + kr] = (__bf16)f.y;
      Bs[(col + 2) * LD + kr] = (__bf16)f.z;
      Bs[(col + 3) * LD + kr] = (__bf16)f.w;
    }
    __syncthreads();

    // ---- fragment loads (contiguous k = hi*16 + s) + 8 WMMAs ----
    Frag16 af[2], bf[4];
#pragma unroll
    for (int fm = 0; fm < 2; ++fm) {
      // A 16x32: lane lo = row, K = hi*16..+15 contiguous
      const bf16_t* p = As + (wm * 32 + fm * 16 + lo) * LD + hi * 16;
      af[fm].q[0] = *(const uint4*)(p);
      af[fm].q[1] = *(const uint4*)(p + 8);
    }
#pragma unroll
    for (int fn = 0; fn < 4; ++fn) {
      // B 32x16: lane lo = col, K = hi*16..+15 contiguous
      const bf16_t* p = Bs + (wn * 64 + fn * 16 + lo) * LD + hi * 16;
      bf[fn].q[0] = *(const uint4*)(p);
      bf[fn].q[1] = *(const uint4*)(p + 8);
    }
#pragma unroll
    for (int fm = 0; fm < 2; ++fm)
#pragma unroll
      for (int fn = 0; fn < 4; ++fn)
        acc[fm][fn].v = wmma_bf16(af[fm].v, bf[fn].v, acc[fm][fn].v);
    __syncthreads();
  }

  // ---- epilogue: C-layout store (row = hi*8+r, col = lo) ----
#pragma unroll
  for (int fm = 0; fm < 2; ++fm)
#pragma unroll
    for (int fn = 0; fn < 4; ++fn)
#pragma unroll
      for (int r = 0; r < 8; ++r) {
        int row = rowBase + wm * 32 + fm * 16 + hi * 8 + r;
        int col = colBase + wn * 64 + fn * 16 + lo;
        Cout[(size_t)row * NTOT + col] = acc[fm][fn].f[r];
      }
}

// ---------------------------------------------------------------------------
// RMSNorm + RoPE + layout conversion.
// Block = 64 threads, one (token, head). Thread t owns the rope pair (2t,2t+1).
// Outputs: Q,K bf16 [B,H,N,HD]  (Q pre-scaled by 1/sqrt(HD))
//          V  bf16 [B,H,HD,N]   (transposed for contiguous P*V B-fragments)
// ---------------------------------------------------------------------------
__global__ __launch_bounds__(64) void norm_rope_kernel(
    const float* __restrict__ qkv, const float* __restrict__ fr,
    const float* __restrict__ fi, const float* __restrict__ qw,
    const float* __restrict__ kw, bf16_t* __restrict__ qb,
    bf16_t* __restrict__ kb, bf16_t* __restrict__ vt) {
  const int bn = blockIdx.x;                  // b*N + n
  const int h  = blockIdx.y;
  const int b  = bn >> 11;                    // N_ = 2048
  const int n  = bn & (N_ - 1);
  const int t  = threadIdx.x;
  const int d0 = 2 * t, d1 = d0 + 1;

  const size_t base = (size_t)bn * 3 * H_ * HD_ + (size_t)h * HD_;
  float q0 = qkv[base + d0],                q1 = qkv[base + d1];
  float k0 = qkv[base + H_ * HD_ + d0],     k1 = qkv[base + H_ * HD_ + d1];
  float v0 = qkv[base + 2 * H_ * HD_ + d0], v1 = qkv[base + 2 * H_ * HD_ + d1];

  float sq = q0 * q0 + q1 * q1;
  float sk = k0 * k0 + k1 * k1;
#pragma unroll
  for (int off = 16; off; off >>= 1) {
    sq += __shfl_xor(sq, off);
    sk += __shfl_xor(sk, off);
  }
  __shared__ float red[4];
  if ((t & 31) == 0) { red[(t >> 5) * 2] = sq; red[(t >> 5) * 2 + 1] = sk; }
  __syncthreads();
  const float qr = rsqrtf((red[0] + red[2]) * (1.0f / HD_) + EPS_);
  const float kr = rsqrtf((red[1] + red[3]) * (1.0f / HD_) + EPS_);

  q0 *= qr * qw[d0]; q1 *= qr * qw[d1];
  k0 *= kr * kw[d0]; k1 *= kr * kw[d1];

  const float frv = fr[(size_t)bn * (HD_ / 2) + t];
  const float fiv = fi[(size_t)bn * (HD_ / 2) + t];
  const float sc  = 0.08838834764831845f;     // 1/sqrt(HD), folded into Q

  const float qo0 = (q0 * frv - q1 * fiv) * sc;
  const float qo1 = (q0 * fiv + q1 * frv) * sc;
  const float ko0 = k0 * frv - k1 * fiv;
  const float ko1 = k0 * fiv + k1 * frv;

  const size_t hb = ((size_t)(b * H_ + h) * N_ + n) * HD_;
  qb[hb + d0] = (__bf16)qo0; qb[hb + d1] = (__bf16)qo1;
  kb[hb + d0] = (__bf16)ko0; kb[hb + d1] = (__bf16)ko1;

  const size_t vtb = (size_t)(b * H_ + h) * HD_;
  vt[(vtb + d0) * N_ + n] = (__bf16)v0;
  vt[(vtb + d1) * N_ + n] = (__bf16)v1;
}

// ---------------------------------------------------------------------------
// Flash attention. Block = 128 threads = 4 waves; wave owns 16 query rows.
// K/V tiles staged in LDS by the Tensor Data Mover (wave 0 issues one
// tensor_load_to_lds per tile; TDM inserts the bank-conflict pad per row).
// Split TENSORcnt waits let the V DMA overlap the S/softmax phase.
// Per 64-key tile: 16 WMMAs for S = Q*K^T, online softmax (f32, shuffle
// reductions matching the C-layout), P re-layout via wave-private LDS strip,
// 16 WMMAs for O += P*V.
// ---------------------------------------------------------------------------
#define KLD 136   // 128 + 8 pad (TDM: 64-dword rows + 4-dword pad)
#define VLD 72    // 64 + 8 pad  (TDM: 32-dword rows + 4-dword pad)

__global__ __launch_bounds__(128) void attn_kernel(
    const bf16_t* __restrict__ qb, const bf16_t* __restrict__ kb,
    const bf16_t* __restrict__ vt, bf16_t* __restrict__ ctx) {
#if USE_TDM
  __shared__ bf16_t Ks[64 * KLD];             // 64 keys x 128 hd (padded)
  __shared__ bf16_t Vs[128 * VLD];            // 128 hd  x 64 keys (padded)
#endif
  __shared__ bf16_t plds[4][16][72];          // P C->A re-layout strips

  const int blk  = blockIdx.x;
  const int tile = blk & 31;                  // N_/64 = 32 q-tiles
  const int h    = (blk >> 5) & (H_ - 1);
  const int b    = blk >> 9;
  const int lane = threadIdx.x & 31;
  const int w    = threadIdx.x >> 5;
  const int lo   = lane & 15;
  const int hi   = lane >> 4;
  const int q0row = tile * 64 + w * 16;

  const bf16_t* qhead = qb + (size_t)(b * H_ + h) * N_ * HD_;
  const bf16_t* khead = kb + (size_t)(b * H_ + h) * N_ * HD_;
  const bf16_t* vhead = vt + (size_t)(b * H_ + h) * HD_ * N_;

  // Q fragments: 16 rows x HD, 4 K-steps of 32
  Frag16 qf[4];
  {
    const bf16_t* qr = qhead + (size_t)(q0row + lo) * HD_ + hi * 16;
#pragma unroll
    for (int kk = 0; kk < 4; ++kk) {
      qf[kk].q[0] = *(const uint4*)(qr + kk * 32);
      qf[kk].q[1] = *(const uint4*)(qr + kk * 32 + 8);
    }
  }

  v8f zero = {};
  F8 O[8];
#pragma unroll
  for (int d = 0; d < 8; ++d) O[d].v = zero;
  float m[8], l[8];
#pragma unroll
  for (int r = 0; r < 8; ++r) { m[r] = -1e30f; l[r] = 0.0f; }

#if USE_TDM
  const unsigned KsAddr = (unsigned)(uintptr_t)(void*)Ks;
  const unsigned VsAddr = (unsigned)(uintptr_t)(void*)Vs;
#endif

  for (int kt = 0; kt < N_; kt += 64) {
#if USE_TDM
    if (w == 0) {
      // K tile: 64 rows x 128 elems, row pitch HD; pad 64dw rows by 4dw
      tdm_load_2d(khead + (size_t)kt * HD_, KsAddr, HD_, 64, HD_, 64,
                  (unsigned long long)HD_, /*interval 64dw*/ 5, /*+4dw*/ 3);
      // V tile: 128 rows x 64 elems, row pitch N; pad 32dw rows by 4dw
      tdm_load_2d(vhead + kt, VsAddr, 64, HD_, 64, HD_,
                  (unsigned long long)N_, /*interval 32dw*/ 4, /*+4dw*/ 3);
      __builtin_amdgcn_s_wait_tensorcnt((short)1);   // K resident, V in flight
    }
    __syncthreads();
#endif

    // ---- S = Q * K^T  (16 x 64) ----
    F8 S[4];
#pragma unroll
    for (int j = 0; j < 4; ++j) {
#if USE_TDM
      const bf16_t* krow = Ks + (j * 16 + lo) * KLD + hi * 16;
#else
      const bf16_t* krow = khead + (size_t)(kt + j * 16 + lo) * HD_ + hi * 16;
#endif
      v8f s = zero;
#pragma unroll
      for (int kk = 0; kk < 4; ++kk) {
        Frag16 kf;
        kf.q[0] = *(const uint4*)(krow + kk * 32);
        kf.q[1] = *(const uint4*)(krow + kk * 32 + 8);
        s = wmma_bf16(qf[kk].v, kf.v, s);
      }
      S[j].v = s;
    }

    // ---- online softmax (each lane owns rows hi*8 + 0..7) ----
    float mn[8], alpha[8];
#pragma unroll
    for (int r = 0; r < 8; ++r) {
      float tmax = fmaxf(fmaxf(S[0].f[r], S[1].f[r]),
                         fmaxf(S[2].f[r], S[3].f[r]));
#pragma unroll
      for (int off = 1; off < 16; off <<= 1)
        tmax = fmaxf(tmax, __shfl_xor(tmax, off, 16));
      mn[r]    = fmaxf(m[r], tmax);
      alpha[r] = __expf(m[r] - mn[r]);
      m[r]     = mn[r];
    }
#pragma unroll
    for (int j = 0; j < 4; ++j)
#pragma unroll
      for (int r = 0; r < 8; ++r)
        S[j].f[r] = __expf(S[j].f[r] - mn[r]);
#pragma unroll
    for (int r = 0; r < 8; ++r) {
      float ts = S[0].f[r] + S[1].f[r] + S[2].f[r] + S[3].f[r];
#pragma unroll
      for (int off = 1; off < 16; off <<= 1)
        ts += __shfl_xor(ts, off, 16);
      l[r] = l[r] * alpha[r] + ts;
    }
#pragma unroll
    for (int d = 0; d < 8; ++d)
#pragma unroll
      for (int r = 0; r < 8; ++r) O[d].f[r] *= alpha[r];

    // ---- re-layout P: C-format -> A-format via wave-private LDS ----
#pragma unroll
    for (int j = 0; j < 4; ++j)
#pragma unroll
      for (int r = 0; r < 8; ++r)
        plds[w][hi * 8 + r][j * 16 + lo] = (__bf16)S[j].f[r];
    asm volatile("s_wait_dscnt 0x0" ::: "memory");

    Frag16 pf[2];
#pragma unroll
    for (int kk2 = 0; kk2 < 2; ++kk2) {
      const bf16_t* p = &plds[w][lo][kk2 * 32 + hi * 16];
      pf[kk2].q[0] = *(const uint4*)(p);
      pf[kk2].q[1] = *(const uint4*)(p + 8);
    }

#if USE_TDM
    if (w == 0) __builtin_amdgcn_s_wait_tensorcnt((short)0);  // V resident
    __syncthreads();
#endif

    // ---- O += P * V ----
#pragma unroll
    for (int d = 0; d < 8; ++d) {
#if USE_TDM
      const bf16_t* vrow = Vs + (d * 16 + lo) * VLD + hi * 16;
#else
      const bf16_t* vrow = vhead + (size_t)(d * 16 + lo) * N_ + kt + hi * 16;
#endif
#pragma unroll
      for (int kk2 = 0; kk2 < 2; ++kk2) {
        Frag16 vf;
        vf.q[0] = *(const uint4*)(vrow + kk2 * 32);
        vf.q[1] = *(const uint4*)(vrow + kk2 * 32 + 8);
        O[d].v = wmma_bf16(pf[kk2].v, vf.v, O[d].v);
      }
    }

#if USE_TDM
    __syncthreads();   // all waves done reading Ks/Vs before next TDM issue
#endif
  }

  // ---- write context [B, N, H*HD] bf16 ----
#pragma unroll
  for (int d = 0; d < 8; ++d)
#pragma unroll
    for (int r = 0; r < 8; ++r) {
      const int tok = q0row + hi * 8 + r;
      const float o = O[d].f[r] / l[r];
      ctx[((size_t)b * N_ + tok) * (H_ * HD_) + h * HD_ + d * 16 + lo] =
          (__bf16)o;
    }
}

// ---------------------------------------------------------------------------
// Launch
// ---------------------------------------------------------------------------
extern "C" void kernel_launch(void* const* d_in, const int* in_sizes, int n_in,
                              void* d_out, int out_size, void* d_ws,
                              size_t ws_size, hipStream_t stream) {
  const float* x    = (const float*)d_in[0];
  const float* fr   = (const float*)d_in[1];
  const float* fi   = (const float*)d_in[2];
  const float* wqkv = (const float*)d_in[3];
  const float* wout = (const float*)d_in[4];
  const float* qw   = (const float*)d_in[5];
  const float* kw   = (const float*)d_in[6];
  float* out = (float*)d_out;

  const size_t M = (size_t)B_ * N_;                       // 4096 tokens
  char* ws = (char*)d_ws;
  float*  qkv = (float*)ws;                               // M * 6144 f32
  size_t  off = M * 3 * H_ * HD_ * sizeof(float);
  bf16_t* qb  = (bf16_t*)(ws + off); off += M * H_ * HD_ * sizeof(bf16_t);
  bf16_t* kb  = (bf16_t*)(ws + off); off += M * H_ * HD_ * sizeof(bf16_t);
  bf16_t* vtb = (bf16_t*)(ws + off); off += M * H_ * HD_ * sizeof(bf16_t);
  bf16_t* ctx = (bf16_t*)(ws + off);

  // 1) QKV projection: [4096,2048] x [2048,6144] -> f32 qkv
  gemm_kernel<3 * H_ * HD_, false>
      <<<dim3((3 * H_ * HD_) / 128, M / 128), 256, 0, stream>>>(x, wqkv, qkv);

  // 2) RMSNorm + RoPE + bf16 conversion / V transpose
  norm_rope_kernel<<<dim3(B_ * N_, H_), 64, 0, stream>>>(qkv, fr, fi, qw, kw,
                                                         qb, kb, vtb);

  // 3) Flash attention -> ctx bf16 [B, N, H*HD]
  attn_kernel<<<dim3(B_ * H_ * (N_ / 64)), 128, 0, stream>>>(qb, kb, vtb, ctx);

  // 4) Output projection: [4096,2048] x [2048,2048] -> f32 d_out
  gemm_kernel<DIM_, true>
      <<<dim3(DIM_ / 128, M / 128), 256, 0, stream>>>(ctx, wout, out);
}
